// Model_1589137899683
// MI455X (gfx1250) — matrix operand order repaired and checked by reading, not compiled, not run on hardware
//
#include <hip/hip_runtime.h>
#include <hip/hip_bf16.h>

// ---------------------------------------------------------------------------
// MI455X (gfx1250, wave32) implementation.
// All dense matmuls go through v_wmma_f32_16x16x32_f16 (f16 in / f32 acc).
// ---------------------------------------------------------------------------

typedef __attribute__((ext_vector_type(16))) _Float16 v16h;
typedef __attribute__((ext_vector_type(8)))  float    v8f;
typedef unsigned int u32;

struct U8 { u32 x[8]; };

// pack two f32 -> packed 2xf16 (v_cvt_pk_rtz_f16_f32)
__device__ __forceinline__ u32 pk2(float x, float y)
{
    auto h = __builtin_amdgcn_cvt_pkrtz(x, y);
    return __builtin_bit_cast(u32, h);
}

// Model constants
#define BSZ   32
#define SEQL  336
#define PREDL 192
#define DMOD  512
#define CIN   64
#define DFF   2048
#define NH    8
#define DH    64
#define YROWS 240
#define YOFF  48
#define OUTN  (BSZ*PREDL*CIN)   // 393216

// ---------------------------------------------------------------------------
// Generic batched GEMM: C = A[MxK] * op(B[KxN]) (+bias)(+relu)(+=C)
// REQUIRES: M % 64 == 0, N % 64 == 0 (true at every call site).
// K may be ragged; only the tail tile takes the guarded slow path.
// 64x64 output tile per 128-thread block (4 waves), BK=32, f16 via LDS.
// A fragment (16x32 f16): lane m=l&15, kh=l>>4; vgpr i ->
//   k = 16*kh + 2i (two consecutive halves -> one b32 read)
// B fragment (32x16 f16): lane n=l&15, kh=l>>4; vgpr i -> k = 16*kh + 2i
// C/D: vgpr i -> m = i + 8*(l>>4), n = l&15
// ---------------------------------------------------------------------------
__global__ __launch_bounds__(128)
void gemm_f16(const float* __restrict__ A, const float* __restrict__ B,
              float* __restrict__ C, const float* __restrict__ bias,
              int M, int N, int K, int lda, int ldb, int ldc,
              long sA1, long sA2, long sB1, long sB2, long sC1, long sC2,
              int nb2, int transB, int biasMode, int accum, int relu)
{
    __shared__ _Float16 sAt[64 * 40];
    __shared__ _Float16 sBt[64 * 40];

    int z  = blockIdx.z;
    int z1 = z / nb2, z2 = z % nb2;
    A += z1 * sA1 + z2 * sA2;
    B += z1 * sB1 + z2 * sB2;
    C += z1 * sC1 + z2 * sC2;

    int m0 = blockIdx.y * 64, n0 = blockIdx.x * 64;
    int tid = threadIdx.x, lane = tid & 31, w = tid >> 5;
    int mw = w * 16;

    v8f zf = {0.f,0.f,0.f,0.f,0.f,0.f,0.f,0.f};
    v8f acc[4];
    acc[0] = zf; acc[1] = zf; acc[2] = zf; acc[3] = zf;

    const u32* uA = (const u32*)sAt;
    const u32* uB = (const u32*)sBt;

    for (int k0 = 0; k0 < K; k0 += 32) {
        __syncthreads();
        if (k0 + 32 <= K) {
            // ---- fast path: unguarded float4 loads, packed f16 stores ----
            #pragma unroll
            for (int it = 0; it < 4; ++it) {
                int f = tid + it * 128;              // 512 float4s: 64r x 8
                int r = f >> 3, c4 = (f & 7) << 2;
                const float4 va = *(const float4*)(A + (long)(m0 + r) * lda + k0 + c4);
                u32* dst = (u32*)&sAt[r * 40 + c4];
                dst[0] = pk2(va.x, va.y);
                dst[1] = pk2(va.z, va.w);
            }
            if (!transB) {
                #pragma unroll
                for (int it = 0; it < 4; ++it) {
                    int f = tid + it * 128;          // 512 float4s: 32k x 16
                    int kk = f >> 4, n4 = (f & 15) << 2;
                    const float4 vb = *(const float4*)(B + (long)(k0 + kk) * ldb + n0 + n4);
                    sBt[(n4 + 0) * 40 + kk] = (_Float16)vb.x;
                    sBt[(n4 + 1) * 40 + kk] = (_Float16)vb.y;
                    sBt[(n4 + 2) * 40 + kk] = (_Float16)vb.z;
                    sBt[(n4 + 3) * 40 + kk] = (_Float16)vb.w;
                }
            } else {
                #pragma unroll
                for (int it = 0; it < 4; ++it) {
                    int f = tid + it * 128;          // 512 float4s: 64n x 8
                    int r = f >> 3, c4 = (f & 7) << 2;
                    const float4 vb = *(const float4*)(B + (long)(n0 + r) * ldb + k0 + c4);
                    u32* dst = (u32*)&sBt[r * 40 + c4];
                    dst[0] = pk2(vb.x, vb.y);
                    dst[1] = pk2(vb.z, vb.w);
                }
            }
        } else {
            // ---- K tail (only K%32 != 0): guarded scalar path ----
            #pragma unroll
            for (int it = 0; it < 16; ++it) {
                int idx = tid + it * 128;
                int r = idx >> 5, c = idx & 31;
                int gk = k0 + c;
                float v = (gk < K) ? A[(long)(m0 + r) * lda + gk] : 0.f;
                sAt[r * 40 + c] = (_Float16)v;
            }
            if (!transB) {
                #pragma unroll
                for (int it = 0; it < 16; ++it) {
                    int idx = tid + it * 128;
                    int kk = idx >> 6, n = idx & 63;
                    int gk = k0 + kk;
                    float v = (gk < K) ? B[(long)gk * ldb + n0 + n] : 0.f;
                    sBt[n * 40 + kk] = (_Float16)v;
                }
            } else {
                #pragma unroll
                for (int it = 0; it < 16; ++it) {
                    int idx = tid + it * 128;
                    int n = idx >> 5, kk = idx & 31;
                    int gk = k0 + kk;
                    float v = (gk < K) ? B[(long)(n0 + n) * ldb + gk] : 0.f;
                    sBt[n * 40 + kk] = (_Float16)v;
                }
            }
        }
        __syncthreads();

        int mr = mw + (lane & 15), kh = lane >> 4;
        U8 au;
        #pragma unroll
        for (int i = 0; i < 8; ++i) {
            int k = (kh << 4) + (i << 1);
            au.x[i] = uA[mr * 20 + (k >> 1)];
        }
        v16h af = __builtin_bit_cast(v16h, au);
        #pragma unroll
        for (int j = 0; j < 4; ++j) {
            int n = j * 16 + (lane & 15);
            U8 bu;
            #pragma unroll
            for (int i = 0; i < 8; ++i) {
                int k = (kh << 4) + (i << 1);
                bu.x[i] = uB[n * 20 + (k >> 1)];
            }
            v16h bf = __builtin_bit_cast(v16h, bu);
            acc[j] = __builtin_amdgcn_wmma_f32_16x16x32_f16(
                false, af, false, bf, (short)0, acc[j], false, false);
        }
    }

    int lane15 = threadIdx.x & 15;
    int hi8 = ((threadIdx.x & 31) >> 4) << 3;
    #pragma unroll
    for (int j = 0; j < 4; ++j) {
        #pragma unroll
        for (int i = 0; i < 8; ++i) {
            int mm = m0 + mw + i + hi8;
            int nn = n0 + j * 16 + lane15;
            long ci = (long)mm * ldc + nn;
            float v = acc[j][i];
            if (accum) v += C[ci];
            if (biasMode == 1) v += bias[nn];
            else if (biasMode == 2) v += bias[mm];
            if (relu) v = v > 0.f ? v : 0.f;
            C[ci] = v;
        }
    }
}

// ---------------------------------------------------------------------------
// RevIN stats + normalize: one block per batch, one thread per channel.
// ---------------------------------------------------------------------------
__global__ __launch_bounds__(64)
void norm_stats(const float* __restrict__ x, float* __restrict__ xe,
                float* __restrict__ means, float* __restrict__ stdev)
{
    int b = blockIdx.x, c = threadIdx.x;
    const float* xb = x + (long)b * SEQL * CIN;
    float s = 0.f;
    for (int t = 0; t < SEQL; ++t) s += xb[t * CIN + c];
    float mu = s / (float)SEQL;
    float v = 0.f;
    for (int t = 0; t < SEQL; ++t) { float d = xb[t * CIN + c] - mu; v += d * d; }
    float sd = sqrtf(v / (float)SEQL + 1e-5f);
    means[b * CIN + c] = mu;
    stdev[b * CIN + c] = sd;
    float inv = 1.f / sd;
    float* xeb = xe + (long)b * SEQL * CIN;
    for (int t = 0; t < SEQL; ++t) xeb[t * CIN + c] = (xb[t * CIN + c] - mu) * inv;
}

// circular pad rows for 3-tap token conv: rows [x[-1], x[0..191], x[0]]
__global__ void build_xpad(const float* __restrict__ cor, float* __restrict__ xp)
{
    long i = (long)blockIdx.x * 256 + threadIdx.x;
    if (i >= (long)BSZ * 194 * CIN) return;
    int c = (int)(i % CIN);
    long t = i / CIN;
    int r = (int)(t % 194);
    int b = (int)(t / 194);
    int src = (r == 0) ? (PREDL - 1) : (r == 193 ? 0 : r - 1);
    xp[i] = cor[((long)b * PREDL + src) * CIN + c];
}

// emb += pos_emb(t,d) + ymark[b,t,:] @ tf_w   (ymark = batch_y_mark[:, -192:, :])
__global__ void embed_add(float* __restrict__ emb, const float* __restrict__ ymark,
                          const float* __restrict__ tfw)
{
    long i = (long)blockIdx.x * 256 + threadIdx.x;
    if (i >= (long)BSZ * PREDL * DMOD) return;
    int d = (int)(i & (DMOD - 1));
    long r = i >> 9;
    int t = (int)(r % PREDL);
    int b = (int)(r / PREDL);
    float freq = expf(-9.210340371976184f * (float)((d >> 1) << 1) / (float)DMOD);
    float ang = (float)t * freq;
    float pe = (d & 1) ? cosf(ang) : sinf(ang);
    const float* mk = ymark + ((long)b * YROWS + YOFF + t) * 4;
    float md = mk[0] * tfw[d] + mk[1] * tfw[DMOD + d]
             + mk[2] * tfw[2 * DMOD + d] + mk[3] * tfw[3 * DMOD + d];
    emb[i] += pe + md;
}

// softmax over last dim (192) with 1/sqrt(dh) scale; one wave per row
__global__ __launch_bounds__(32)
void softmax192(float* __restrict__ s)
{
    float* r = s + (long)blockIdx.x * PREDL;
    int lane = threadIdx.x;
    float v[6];
    float mx = -3.4e38f;
    #pragma unroll
    for (int i = 0; i < 6; ++i) { v[i] = r[lane + i * 32] * 0.125f; mx = fmaxf(mx, v[i]); }
    #pragma unroll
    for (int o = 16; o; o >>= 1) mx = fmaxf(mx, __shfl_xor(mx, o));
    float sum = 0.f;
    #pragma unroll
    for (int i = 0; i < 6; ++i) { v[i] = expf(v[i] - mx); sum += v[i]; }
    #pragma unroll
    for (int o = 16; o; o >>= 1) sum += __shfl_xor(sum, o);
    float inv = 1.f / sum;
    #pragma unroll
    for (int i = 0; i < 6; ++i) r[lane + i * 32] = v[i] * inv;
}

// out = LayerNorm(x + res) * g + b ; one wave per 512-row
__global__ __launch_bounds__(32)
void add_ln(const float* __restrict__ x, const float* __restrict__ res,
            const float* __restrict__ g, const float* __restrict__ be,
            float* __restrict__ out)
{
    long row = blockIdx.x;
    int lane = threadIdx.x;
    const float* xr = x + row * DMOD;
    const float* rr = res + row * DMOD;
    float v[16];
    float s = 0.f;
    #pragma unroll
    for (int i = 0; i < 16; ++i) { v[i] = xr[lane + i * 32] + rr[lane + i * 32]; s += v[i]; }
    #pragma unroll
    for (int o = 16; o; o >>= 1) s += __shfl_xor(s, o);
    float mu = s * (1.f / (float)DMOD);
    float q = 0.f;
    #pragma unroll
    for (int i = 0; i < 16; ++i) { float d = v[i] - mu; q += d * d; }
    #pragma unroll
    for (int o = 16; o; o >>= 1) q += __shfl_xor(q, o);
    float inv = rsqrtf(q * (1.f / (float)DMOD) + 1e-5f);
    float* orow = out + row * DMOD;
    #pragma unroll
    for (int i = 0; i < 16; ++i) {
        int cc = lane + i * 32;
        orow[cc] = (v[i] - mu) * inv * g[cc] + be[cc];
    }
}

// X = concat([corrupt, cond], -1)  -> (B*P, 128)
__global__ void build_x(const float* __restrict__ cor, const float* __restrict__ cond,
                        float* __restrict__ X)
{
    long i = (long)blockIdx.x * 256 + threadIdx.x;
    if (i >= (long)BSZ * PREDL * 128) return;
    int c = (int)(i & 127);
    long r = i >> 7;
    X[i] = (c < CIN) ? cor[r * CIN + c] : cond[r * CIN + (c - CIN)];
}

// ---------------------------------------------------------------------------
// GRU scan step (only kernel on the serial hx chain): per block, 64 hidden
// columns j0.. across the 3 gates.  gh = hx @ whh^T (WMMA, K=512) -> gates ->
// h_new + analytic dgi written into row (b*192+p) of the full DGI matrix.
// The gradient GEMM g = DGI @ wih[:, :64] is hoisted OUT of the scan (it is
// not on the recurrence) and done afterwards as one 6144x64x1536 WMMA GEMM.
// ---------------------------------------------------------------------------
__global__ __launch_bounds__(128)
void gru_step1(const float* __restrict__ hx_in, const float* __restrict__ whh,
               const float* __restrict__ bhh, const float* __restrict__ GI,
               const float* __restrict__ score_w, float* __restrict__ hx_out,
               float* __restrict__ dgi, int p)
{
    __shared__ _Float16 sH[32 * 512];
    __shared__ _Float16 sB[64 * 40];
    __shared__ float    sGH[3 * 32 * 64];

    int tid = threadIdx.x, lane = tid & 31, w = tid >> 5;
    int j0 = blockIdx.x * 64;

    // stage hx (32x512) as f16, vectorized
    #pragma unroll
    for (int it = 0; it < 32; ++it) {
        int f = tid + it * 128;                  // 4096 float4s
        const float4 v = *(const float4*)(hx_in + (long)f * 4);
        u32* dst = (u32*)&sH[f * 4];
        dst[0] = pk2(v.x, v.y);
        dst[1] = pk2(v.z, v.w);
    }
    __syncthreads();

    const u32* uH = (const u32*)sH;
    const u32* uB = (const u32*)sB;
    int mw = (w & 1) * 16;
    int nA = (w >> 1) * 32;
    v8f zf = {0.f,0.f,0.f,0.f,0.f,0.f,0.f,0.f};

    for (int gate = 0; gate < 3; ++gate) {
        int jb = gate * DMOD + j0;
        v8f a0 = zf, a1 = zf;
        for (int k0 = 0; k0 < DMOD; k0 += 32) {
            __syncthreads();
            #pragma unroll
            for (int it = 0; it < 4; ++it) {
                int f = tid + it * 128;          // 512 float4s: 64col x 8
                int col = f >> 3, c4 = (f & 7) << 2;
                const float4 v = *(const float4*)(whh + (long)(jb + col) * DMOD + k0 + c4);
                u32* dst = (u32*)&sB[col * 40 + c4];
                dst[0] = pk2(v.x, v.y);
                dst[1] = pk2(v.z, v.w);
            }
            __syncthreads();
            int mr = mw + (lane & 15), kh = lane >> 4;
            U8 au;
            #pragma unroll
            for (int i = 0; i < 8; ++i) {
                int k = k0 + (kh << 4) + (i << 1);
                au.x[i] = uH[mr * 256 + (k >> 1)];
            }
            v16h af = __builtin_bit_cast(v16h, au);
            U8 bu;
            #pragma unroll
            for (int i = 0; i < 8; ++i) {
                int k = (kh << 4) + (i << 1);
                bu.x[i] = uB[(nA + (lane & 15)) * 20 + (k >> 1)];
            }
            v16h bf0 = __builtin_bit_cast(v16h, bu);
            a0 = __builtin_amdgcn_wmma_f32_16x16x32_f16(false, af, false, bf0, (short)0, a0, false, false);
            #pragma unroll
            for (int i = 0; i < 8; ++i) {
                int k = (kh << 4) + (i << 1);
                bu.x[i] = uB[(nA + 16 + (lane & 15)) * 20 + (k >> 1)];
            }
            v16h bf1 = __builtin_bit_cast(v16h, bu);
            a1 = __builtin_amdgcn_wmma_f32_16x16x32_f16(false, af, false, bf1, (short)0, a1, false, false);
        }
        int hi8 = (lane >> 4) << 3;
        #pragma unroll
        for (int i = 0; i < 8; ++i) {
            int m = mw + i + hi8;
            sGH[gate * 2048 + m * 64 + nA + (lane & 15)]      = a0[i];
            sGH[gate * 2048 + m * 64 + nA + 16 + (lane & 15)] = a1[i];
        }
    }
    __syncthreads();

    for (int e = tid; e < 2048; e += 128) {
        int b = e >> 6, jj = e & 63, j = j0 + jj;
        long row = (long)b * PREDL + p;
        float gir = GI[row * 1536 + j];
        float giz = GI[row * 1536 + 512 + j];
        float gin = GI[row * 1536 + 1024 + j];
        float ghr = sGH[0 * 2048 + b * 64 + jj] + bhh[j];
        float ghz = sGH[1 * 2048 + b * 64 + jj] + bhh[512 + j];
        float ghn = sGH[2 * 2048 + b * 64 + jj] + bhh[1024 + j];
        float r = 1.f / (1.f + expf(-(gir + ghr)));
        float z = 1.f / (1.f + expf(-(giz + ghz)));
        float n = tanhf(gin + r * ghn);
        float h = hx_in[b * DMOD + j];
        hx_out[b * DMOD + j] = (1.f - z) * n + z * h;
        float dEdh = 0.1f * score_w[j];          // COF * score_w
        float dn_  = dEdh * (1.f - z);
        float dinn = dn_ * (1.f - n * n);
        float dir_ = dinn * ghn * r * (1.f - r);
        float dz   = dEdh * (h - n);
        float diz  = dz * z * (1.f - z);
        dgi[row * 1536 + j]        = dir_;
        dgi[row * 1536 + 512 + j]  = diz;
        dgi[row * 1536 + 1024 + j] = dinn;
    }
}

// pre = corrupt - alpha * G   (Langevin update, done once after the scan)
__global__ void langevin_update(const float* __restrict__ cor,
                                const float* __restrict__ G,
                                const float* __restrict__ alpha,
                                float* __restrict__ pre)
{
    long i = (long)blockIdx.x * 256 + threadIdx.x;
    if (i < (long)OUTN) pre[i] = cor[i] - alpha[0] * G[i];
}

// out_dn = pre*stdev + means ; per-block partial sums of |out_dn - y| (deterministic)
__global__ __launch_bounds__(256)
void outdn_loss(const float* __restrict__ pre, const float* __restrict__ means,
                const float* __restrict__ stdev, const float* __restrict__ y,
                float* __restrict__ out, float* __restrict__ part, int writeOut)
{
    long i = (long)blockIdx.x * 256 + threadIdx.x;
    float d = 0.f;
    if (i < (long)OUTN) {
        int c = (int)(i & (CIN - 1));
        long r = i >> 6;
        int p = (int)(r % PREDL);
        int b = (int)(r / PREDL);
        float val = pre[i] * stdev[b * CIN + c] + means[b * CIN + c];
        float yy = y[((long)b * YROWS + YOFF + p) * CIN + c];
        d = fabsf(val - yy);
        if (writeOut) out[i] = val;
    }
    __shared__ float red[256];
    red[threadIdx.x] = d;
    __syncthreads();
    for (int st = 128; st > 0; st >>= 1) {
        if (threadIdx.x < st) red[threadIdx.x] += red[threadIdx.x + st];
        __syncthreads();
    }
    if (threadIdx.x == 0) part[blockIdx.x] = red[0];
}

__global__ __launch_bounds__(256)
void finalize_loss(const float* __restrict__ part, float* __restrict__ out)
{
    __shared__ float red[256];
    float s = 0.f;
    for (int i = threadIdx.x; i < 3072; i += 256) s += part[i];
    red[threadIdx.x] = s;
    __syncthreads();
    for (int st = 128; st > 0; st >>= 1) {
        if (threadIdx.x < st) red[threadIdx.x] += red[threadIdx.x + st];
        __syncthreads();
    }
    if (threadIdx.x == 0) out[OUTN] = red[0] * (1.f / (float)OUTN);
}

// ---------------------------------------------------------------------------
static inline void launch_gemm(hipStream_t st, const float* A, const float* B,
                               float* C, const float* bias,
                               int M, int N, int K, int lda, int ldb, int ldc,
                               long sA1, long sA2, long sB1, long sB2,
                               long sC1, long sC2, int nb1, int nb2,
                               int transB, int biasMode, int accum, int relu)
{
    dim3 g((N + 63) / 64, (M + 63) / 64, nb1 * nb2);
    gemm_f16<<<g, 128, 0, st>>>(A, B, C, bias, M, N, K, lda, ldb, ldc,
                                sA1, sA2, sB1, sB2, sC1, sC2, nb2,
                                transB, biasMode, accum, relu);
}

extern "C" void kernel_launch(void* const* d_in, const int* in_sizes, int n_in,
                              void* d_out, int out_size, void* d_ws, size_t ws_size,
                              hipStream_t stream)
{
    (void)in_sizes; (void)n_in; (void)out_size; (void)ws_size;

    const float* x_enc        = (const float*)d_in[0];
    const float* batch_y      = (const float*)d_in[2];
    const float* batch_y_mark = (const float*)d_in[3];
    const float* hx_init      = (const float*)d_in[4];
    const float* token_conv_w = (const float*)d_in[5];
    const float* timefeat_w   = (const float*)d_in[6];
    const float* Wq   = (const float*)d_in[7];
    const float* Wk   = (const float*)d_in[8];
    const float* Wv   = (const float*)d_in[9];
    const float* Wo   = (const float*)d_in[10];
    const float* bq   = (const float*)d_in[11];
    const float* bk   = (const float*)d_in[12];
    const float* bv   = (const float*)d_in[13];
    const float* bo   = (const float*)d_in[14];
    const float* ln1g = (const float*)d_in[15];
    const float* ln1b = (const float*)d_in[16];
    const float* W1   = (const float*)d_in[17];
    const float* b1   = (const float*)d_in[18];
    const float* W2   = (const float*)d_in[19];
    const float* b2   = (const float*)d_in[20];
    const float* ln2g = (const float*)d_in[21];
    const float* ln2b = (const float*)d_in[22];
    const float* pl1w = (const float*)d_in[23];
    const float* pl1b = (const float*)d_in[24];
    const float* pl2w = (const float*)d_in[25];
    const float* pl2b = (const float*)d_in[26];
    const float* projw = (const float*)d_in[27];
    const float* projb = (const float*)d_in[28];
    const float* gwih = (const float*)d_in[29];
    const float* gwhh = (const float*)d_in[30];
    const float* gbih = (const float*)d_in[31];
    const float* gbhh = (const float*)d_in[32];
    const float* scw  = (const float*)d_in[33];
    const float* alpha = (const float*)d_in[34];
    float* out = (float*)d_out;

    float* ws = (float*)d_ws;
    size_t off = 0;
    auto alloc = [&](size_t n) { float* p = ws + off; off += n; return p; };
    float* xe      = alloc((size_t)BSZ * SEQL * CIN);      // 688128
    float* means   = alloc(2048);
    float* stdev   = alloc(2048);
    float* pre     = alloc(OUTN);                          // 393216
    float* corrupt = alloc(OUTN);
    float* xpad    = alloc((size_t)BSZ * 194 * CIN);       // 397312
    float* emb     = alloc((size_t)BSZ * PREDL * DMOD);    // 3145728
    float* qb      = alloc((size_t)BSZ * PREDL * DMOD);
    float* kb      = alloc((size_t)BSZ * PREDL * DMOD);
    float* vb      = alloc((size_t)BSZ * PREDL * DMOD);
    float* ctx     = alloc((size_t)BSZ * PREDL * DMOD);
    float* scores  = alloc((size_t)BSZ * NH * PREDL * PREDL); // 9437184; reused as GI
    float* ff1     = alloc((size_t)BSZ * PREDL * DFF);     // 12582912; reused as DGI
    float* cond    = alloc(OUTN);                          // reused as G after GI built
    float* Xb      = alloc((size_t)BSZ * PREDL * 128);     // 786432
    float* hx0     = alloc((size_t)BSZ * DMOD);
    float* hx1     = alloc((size_t)BSZ * DMOD);
    float* part    = alloc(3072);

    const long TD = (long)PREDL * DMOD;   // 98304
    const long SS = (long)PREDL * PREDL;  // 36864

    // 1) RevIN normalize
    norm_stats<<<BSZ, CIN, 0, stream>>>(x_enc, xe, means, stdev);

    // 2) pre = pl1_w @ xe[b] + pl1_b  (per batch)
    launch_gemm(stream, pl1w, xe, pre, pl1b, PREDL, CIN, SEQL, SEQL, CIN, CIN,
                0, 0, (long)SEQL * CIN, 0, (long)PREDL * CIN, 0, BSZ, 1, 0, 2, 0, 0);

    for (int s = 0; s < 2; ++s) {
        // corrupt = pl2_w @ pre[b] + pl2_b
        launch_gemm(stream, pl2w, pre, corrupt, pl2b, PREDL, CIN, PREDL,
                    PREDL, CIN, CIN, 0, 0, (long)PREDL * CIN, 0,
                    (long)PREDL * CIN, 0, BSZ, 1, 0, 2, 0, 0);

        // token conv embedding
        build_xpad<<<(BSZ * 194 * CIN + 255) / 256, 256, 0, stream>>>(corrupt, xpad);
        for (int tap = 0; tap < 3; ++tap) {
            launch_gemm(stream, xpad + tap * CIN, token_conv_w + (long)tap * CIN * DMOD,
                        emb, nullptr, PREDL, DMOD, CIN, CIN, DMOD, DMOD,
                        (long)194 * CIN, 0, 0, 0, TD, 0, BSZ, 1, 0, 0, tap > 0 ? 1 : 0, 0);
        }
        embed_add<<<(int)((TD * BSZ + 255) / 256), 256, 0, stream>>>(emb, batch_y_mark, timefeat_w);

        // transformer blocks
        for (int l = 0; l < 2; ++l) {
            long wo = (long)l * DMOD * DMOD, bofs = (long)l * DMOD;
            long w1o = (long)l * DMOD * DFF, b1o = (long)l * DFF;
            long w2o = (long)l * DFF * DMOD;
            int Mfull = BSZ * PREDL;
            launch_gemm(stream, emb, Wq + wo, qb, bq + bofs, Mfull, DMOD, DMOD,
                        DMOD, DMOD, DMOD, 0, 0, 0, 0, 0, 0, 1, 1, 0, 1, 0, 0);
            launch_gemm(stream, emb, Wk + wo, kb, bk + bofs, Mfull, DMOD, DMOD,
                        DMOD, DMOD, DMOD, 0, 0, 0, 0, 0, 0, 1, 1, 0, 1, 0, 0);
            launch_gemm(stream, emb, Wv + wo, vb, bv + bofs, Mfull, DMOD, DMOD,
                        DMOD, DMOD, DMOD, 0, 0, 0, 0, 0, 0, 1, 1, 0, 1, 0, 0);
            // scores[b,h] = q[b,:,h,:] @ k[b,:,h,:]^T
            launch_gemm(stream, qb, kb, scores, nullptr, PREDL, PREDL, DH,
                        DMOD, DMOD, PREDL, TD, DH, TD, DH, (long)NH * SS, SS,
                        BSZ, NH, 1, 0, 0, 0);
            softmax192<<<BSZ * NH * PREDL, 32, 0, stream>>>(scores);
            // ctx[b,:,h,:] = att @ v
            launch_gemm(stream, scores, vb, ctx, nullptr, PREDL, DH, PREDL,
                        PREDL, DMOD, DMOD, (long)NH * SS, SS, TD, DH, TD, DH,
                        BSZ, NH, 0, 0, 0, 0);
            // o = ctx @ Wo + bo  (into qb)
            launch_gemm(stream, ctx, Wo + wo, qb, bo + bofs, Mfull, DMOD, DMOD,
                        DMOD, DMOD, DMOD, 0, 0, 0, 0, 0, 0, 1, 1, 0, 1, 0, 0);
            add_ln<<<Mfull, 32, 0, stream>>>(emb, qb, ln1g + bofs, ln1b + bofs, kb);
            launch_gemm(stream, kb, W1 + w1o, ff1, b1 + b1o, Mfull, DFF, DMOD,
                        DMOD, DFF, DFF, 0, 0, 0, 0, 0, 0, 1, 1, 0, 1, 0, 1);
            launch_gemm(stream, ff1, W2 + w2o, vb, b2 + bofs, Mfull, DMOD, DFF,
                        DFF, DMOD, DMOD, 0, 0, 0, 0, 0, 0, 1, 1, 0, 1, 0, 0);
            add_ln<<<Mfull, 32, 0, stream>>>(kb, vb, ln2g + bofs, ln2b + bofs, emb);
        }

        // cond = emb @ proj_w^T + proj_b
        launch_gemm(stream, emb, projw, cond, projb, BSZ * PREDL, CIN, DMOD,
                    DMOD, DMOD, CIN, 0, 0, 0, 0, 0, 0, 1, 1, 1, 1, 0, 0);

        // GI = [corrupt|cond] @ wih^T + bih  (hoisted out of the scan)
        build_x<<<(BSZ * PREDL * 128 + 255) / 256, 256, 0, stream>>>(corrupt, cond, Xb);
        launch_gemm(stream, Xb, gwih, scores /*GI*/, gbih, BSZ * PREDL, 1536, 128,
                    128, 128, 1536, 0, 0, 0, 0, 0, 0, 1, 1, 1, 1, 0, 0);

        // Langevin scan: only the hx recurrence is sequential (192 launches).
        // dgi for every step accumulates into DGI (6144 x 1536, reusing ff1).
        float* DGI = ff1;
        for (int p = 0; p < PREDL; ++p) {
            const float* hin = (p == 0) ? (hx_init + (long)s * BSZ * DMOD)
                                        : ((p & 1) ? hx1 : hx0);
            float* hout = (p & 1) ? hx0 : hx1;
            gru_step1<<<8, 128, 0, stream>>>(hin, gwhh, gbhh, scores, scw, hout, DGI, p);
        }
        // G = DGI @ wih[:, :64]  (one full-size WMMA GEMM; G reuses cond)
        launch_gemm(stream, DGI, gwih, cond, nullptr, BSZ * PREDL, CIN, 1536,
                    1536, 128, CIN, 0, 0, 0, 0, 0, 0, 1, 1, 0, 0, 0, 0);
        // pre = corrupt - alpha * G
        langevin_update<<<(OUTN + 255) / 256, 256, 0, stream>>>(corrupt, cond, alpha, pre);

        // denorm + L1 partials (write final output on last step)
        outdn_loss<<<(OUTN + 255) / 256, 256, 0, stream>>>(
            pre, means, stdev, batch_y, out, part + (long)s * 1536, s == 1 ? 1 : 0);
    }

    finalize_loss<<<1, 256, 0, stream>>>(part, out);
}
